// Decoder_59072980189506
// MI455X (gfx1250) — matrix operand order repaired and checked by reading, not compiled, not run on hardware
//
#include <hip/hip_runtime.h>
#include <hip/hip_bf16.h>
#include <cstdint>

typedef __attribute__((ext_vector_type(16))) _Float16 v16h;
typedef __attribute__((ext_vector_type(8)))  float    v8f;

#define B_SZ 64
#define TENC 512
#define TDEC 400
#define DIN  160

union AFrag { v16h v; _Float16 e[16]; };
union CFrag { v8f  v; float    e[8];  };

// ---------------------------------------------------------------------------
// Packed A-fragment index for a 16xK f16 activation tile in LDS.
// A-fragment layout (ISA 7.12.2, 16-bit A 16x32): m = lane&15,
//   lanes<16: K = {0..7,16..23}; lanes>=16: K = {8..15,24..31}.
// Element (m,k): ks=k>>5, r5=k&31, h=(r5>>3)&1, g=r5>>4, o=r5&7
//   -> lane = m + 16h, i = 8g + o, idx = (ks*32 + lane)*16 + i
// ---------------------------------------------------------------------------
__device__ inline int packA(int m, int k)
{
    int ks = k >> 5;
    int r5 = k & 31;
    int h  = (r5 >> 3) & 1;
    int g  = r5 >> 4;
    int o  = r5 & 7;
    return (((ks << 5) + (h << 4) + m) << 4) + (g << 3) + o;
}

// ---------------------------------------------------------------------------
// Weight packing: KxN f32 row-major -> per-fragment f16 so each lane loads one
// contiguous v16h (32B) per WMMA K-step.
// Packed index = ((tn*nK + ks)*32 + lane)*16 + i
// B-fragment layout (ISA 7.12.2, 16-bit B 32x16):
//   n = tn*16 + (lane&15);  k = ks*32 + (lane>=16 ? 8 : 0) + (i>=8 ? 16 : 0) + (i&7)
// ---------------------------------------------------------------------------
__global__ void pack_kernel(const float* __restrict__ W, _Float16* __restrict__ dst,
                            int K, int N)
{
    int idx = blockIdx.x * blockDim.x + threadIdx.x;
    int total = K * N;
    if (idx >= total) return;
    int i    = idx & 15;
    int lane = (idx >> 4) & 31;
    int t2   = idx >> 9;
    int nK   = K >> 5;
    int ks   = t2 % nK;
    int tn   = t2 / nK;
    int k = (ks << 5) + ((lane >> 4) << 3) + ((i >> 3) << 4) + (i & 7);
    int n = (tn << 4) + (lane & 15);
    dst[idx] = (_Float16)W[(size_t)k * N + n];
}

// ---------------------------------------------------------------------------
// Cooperative 16xN GEMM: C(16,N) = A(16,K, pre-packed f16 fragments in LDS)
//                               @ (packed-f16 B) (+bias).
// Each wave owns n-tiles round robin; f32 accumulate via v_wmma_f32_16x16x32_f16.
// A fragment = one 32B LDS read per K-step; B fragment = one 32B global read.
// C/D f32 16x16: n = lane&15, m = j + 8*(lane>=16).
// ---------------------------------------------------------------------------
__device__ inline void gemm16(const _Float16* __restrict__ sAh, int K,
                              const _Float16* __restrict__ Bp, int N,
                              const float* __restrict__ bias,
                              float* __restrict__ out, size_t out_stride,
                              int relu, int wave, int nwaves, int lane)
{
    const int nK     = K >> 5;
    const int nTiles = N >> 4;
    const int col    = lane & 15;
    const int half8  = (lane >> 4) << 3;          // 0 or 8
    for (int tn = wave; tn < nTiles; tn += nwaves) {
        CFrag c;
#pragma unroll
        for (int j = 0; j < 8; ++j) c.e[j] = 0.0f;
        const _Float16* bp = Bp + (size_t)tn * nK * 512 + (size_t)lane * 16;
        const _Float16* ap = sAh + (size_t)lane * 16;
        for (int ks = 0; ks < nK; ++ks) {
            v16h a = *(const v16h*)ap;
            ap += 512;                             // 32 lanes * 16 halves
            v16h b = *(const v16h*)bp;
            bp += 512;
            c.v = __builtin_amdgcn_wmma_f32_16x16x32_f16(
                false, a, false, b, (short)0, c.v, false, false);
        }
#pragma unroll
        for (int j = 0; j < 8; ++j) {
            int m = half8 + j;
            int n = (tn << 4) + col;
            float v = c.e[j];
            if (bias) v += bias[n];
            if (relu) v = v > 0.0f ? v : 0.0f;
            out[(size_t)m * out_stride + n] = v;
        }
    }
}

// ---------------------------------------------------------------------------
// One-shot parallel precompute: pm(B*TENC, 256) = enc(B*TENC, 256) @ mem_W.
// 32768 rows -> 2048 M-tiles x 16 N-tiles; one wave per output tile.
// A built from global f32 on the fly (one-shot, compiler pipelines the loads).
// ---------------------------------------------------------------------------
__global__ __launch_bounds__(256) void pm_kernel(const float* __restrict__ enc,
                                                 const _Float16* __restrict__ memWp,
                                                 float* __restrict__ pm)
{
    int lane  = threadIdx.x & 31;
    int wave  = threadIdx.x >> 5;
    int gw    = blockIdx.x * 8 + wave;     // 0..32767
    int mtile = gw >> 4;                   // 0..2047
    int tn    = gw & 15;                   // 0..15
    int col   = lane & 15;
    int half8 = (lane >> 4) << 3;

    CFrag c;
#pragma unroll
    for (int j = 0; j < 8; ++j) c.e[j] = 0.0f;

    const float* arow = enc + ((size_t)mtile * 16 + (lane & 15)) * 256;
    const _Float16* bp = memWp + (size_t)tn * 8 * 512 + (size_t)lane * 16;
    for (int ks = 0; ks < 8; ++ks) {
        AFrag a;
        const float* ap = arow + (ks << 5);
#pragma unroll
        for (int i = 0; i < 8; ++i) a.e[i]     = (_Float16)ap[half8 + i];
#pragma unroll
        for (int i = 0; i < 8; ++i) a.e[8 + i] = (_Float16)ap[16 + half8 + i];
        v16h b = *(const v16h*)bp;
        bp += 512;
        c.v = __builtin_amdgcn_wmma_f32_16x16x32_f16(
            false, a.v, false, b, (short)0, c.v, false, false);
    }
#pragma unroll
    for (int j = 0; j < 8; ++j) {
        int m = mtile * 16 + half8 + j;
        int n = tn * 16 + col;
        pm[(size_t)m * 256 + n] = c.e[j];
    }
}

__device__ inline void gru_combine(const float* __restrict__ gi, const float* __restrict__ gh,
                                   float* __restrict__ h, int b0, int tid)
{
    for (int idx = tid; idx < 16 * 256; idx += 1024) {
        int r = idx >> 8, j = idx & 255;
        size_t gb = (size_t)(b0 + r) * 768;
        size_t hb = (size_t)(b0 + r) * 256 + j;
        float rg = gi[gb + j]       + gh[gb + j];
        float zg = gi[gb + 256 + j] + gh[gb + 256 + j];
        float rr = 1.f / (1.f + __expf(-rg));
        float zz = 1.f / (1.f + __expf(-zg));
        float nn = tanhf(gi[gb + 512 + j] + rr * gh[gb + 512 + j]);
        h[hb] = (1.f - zz) * nn + zz * h[hb];
    }
}

struct DecP {
    const float* enc; const float* pm; const float* inp; const int* memlen;
    const float* vW;
    const _Float16 *preW1, *preW2, *qW, *attWi, *attWh, *projW, *d1Wi, *d1Wh, *d2Wi, *d2Wh, *melW;
    const float *preB1, *preB2, *attBi, *attBh, *projB, *d1Bi, *d1Bh, *d2Bi, *d2Bh, *melB;
    const float *gateW, *gateB;
    float *attH, *h1, *h2, *curAtt, *gi, *gh, *q, *t256, *t128, *decin, *dcur;
    float *outMel, *outAlign, *outGate;
};

// ---------------------------------------------------------------------------
// Persistent recurrent kernel: 4 blocks x 1024 threads (32 wave32 waves).
// Block k owns batch rows [16k, 16k+16) through all 400 steps; the 16-row
// batch tile maps exactly onto the WMMA M=16 dimension. Activations are
// staged into LDS already converted to f16 in WMMA A-fragment order.
// ---------------------------------------------------------------------------
__global__ __launch_bounds__(1024) void decoder_kernel(DecP p)
{
    __shared__ __align__(32) _Float16 sAh[16 * 512];   // 16 KB packed A stage
    const int tid  = threadIdx.x;
    const int lane = tid & 31;
    const int wave = tid >> 5;
    const int NW   = 32;
    const int b0   = blockIdx.x << 4;

    // zero initial states for this block's rows
    for (int idx = tid; idx < 16 * 256; idx += 1024) {
        int r = idx >> 8, c = idx & 255;
        size_t o = (size_t)(b0 + r) * 256 + c;
        p.attH[o] = 0.f; p.h1[o] = 0.f; p.h2[o] = 0.f; p.curAtt[o] = 0.f;
    }
    __syncthreads();

    for (int t = 0; t < TDEC; ++t) {
        // ---------------- prenet ----------------
        for (int idx = tid; idx < 16 * DIN; idx += 1024) {
            int r = idx / DIN, c = idx - r * DIN;
            float v = 0.f;
            if (t > 0) v = p.inp[((size_t)(b0 + r) * TDEC + (t - 1)) * DIN + c];
            sAh[packA(r, c)] = (_Float16)v;
        }
        __syncthreads();
        gemm16(sAh, DIN, p.preW1, 256, p.preB1, p.t256 + (size_t)b0 * 256, 256, 1, wave, NW, lane);
        __syncthreads();
        for (int idx = tid; idx < 16 * 256; idx += 1024)
            sAh[packA(idx >> 8, idx & 255)] = (_Float16)p.t256[(size_t)b0 * 256 + idx];
        __syncthreads();
        gemm16(sAh, 256, p.preW2, 128, p.preB2, p.t128 + (size_t)b0 * 128, 128, 1, wave, NW, lane);
        __syncthreads();

        // ---------------- attention GRU ----------------
        for (int idx = tid; idx < 16 * 384; idx += 1024) {
            int r = idx / 384, c = idx - r * 384;
            float v = (c < 128) ? p.t128[(size_t)(b0 + r) * 128 + c]
                                : p.curAtt[(size_t)(b0 + r) * 256 + (c - 128)];
            sAh[packA(r, c)] = (_Float16)v;
        }
        __syncthreads();
        gemm16(sAh, 384, p.attWi, 768, p.attBi, p.gi + (size_t)b0 * 768, 768, 0, wave, NW, lane);
        __syncthreads();
        for (int idx = tid; idx < 16 * 256; idx += 1024)
            sAh[packA(idx >> 8, idx & 255)] = (_Float16)p.attH[(size_t)b0 * 256 + idx];
        __syncthreads();
        gemm16(sAh, 256, p.attWh, 768, p.attBh, p.gh + (size_t)b0 * 768, 768, 0, wave, NW, lane);
        __syncthreads();
        gru_combine(p.gi, p.gh, p.attH, b0, tid);
        __syncthreads();

        // ---------------- attention query ----------------
        for (int idx = tid; idx < 16 * 256; idx += 1024)
            sAh[packA(idx >> 8, idx & 255)] = (_Float16)p.attH[(size_t)b0 * 256 + idx];
        __syncthreads();
        gemm16(sAh, 256, p.qW, 256, nullptr, p.q + (size_t)b0 * 256, 256, 0, wave, NW, lane);
        __syncthreads();

        // ---------------- score (into alignment output, in place) ----------------
        for (int idx = tid; idx < 16 * TENC; idx += 1024) {
            int r = idx >> 9, te = idx & (TENC - 1);
            int b = b0 + r;
            float s = -1e9f;
            if (te < p.memlen[b]) {
                const float* pmr = p.pm + ((size_t)b * TENC + te) * 256;
                // prefetch the pm row this thread handles on its next iteration
                if (r + 2 < 16)
                    __builtin_prefetch(p.pm + ((size_t)(b + 2) * TENC + te) * 256, 0, 1);
                const float* qr  = p.q + (size_t)b * 256;
                float acc = 0.f;
                for (int d = 0; d < 256; ++d) acc += p.vW[d] * tanhf(pmr[d] + qr[d]);
                s = acc;
            }
            p.outAlign[((size_t)b * TDEC + t) * TENC + te] = s;
        }
        __syncthreads();
        // ---------------- softmax (one wave per batch row) ----------------
        if (wave < 16) {
            int b = b0 + wave;
            float* srow = p.outAlign + ((size_t)b * TDEC + t) * TENC;
            float mx = -1e30f;
            for (int te = lane; te < TENC; te += 32) mx = fmaxf(mx, srow[te]);
            for (int off = 16; off > 0; off >>= 1) mx = fmaxf(mx, __shfl_xor(mx, off, 32));
            float sum = 0.f;
            for (int te = lane; te < TENC; te += 32) { float e = __expf(srow[te] - mx); srow[te] = e; sum += e; }
            for (int off = 16; off > 0; off >>= 1) sum += __shfl_xor(sum, off, 32);
            float inv = 1.f / sum;
            for (int te = lane; te < TENC; te += 32) srow[te] *= inv;
        }
        __syncthreads();
        // ---------------- context ----------------
        for (int idx = tid; idx < 16 * 256; idx += 1024) {
            int r = idx >> 8, d = idx & 255;
            int b = b0 + r;
            const float* al = p.outAlign + ((size_t)b * TDEC + t) * TENC;
            const float* er = p.enc + (size_t)b * TENC * 256 + d;
            float acc = 0.f;
            for (int te = 0; te < TENC; ++te) acc += al[te] * er[(size_t)te * 256];
            p.curAtt[(size_t)b * 256 + d] = acc;
        }
        __syncthreads();

        // ---------------- project to decoder input ----------------
        for (int idx = tid; idx < 16 * 512; idx += 1024) {
            int r = idx >> 9, c = idx & 511;
            float v = (c < 256) ? p.attH[(size_t)(b0 + r) * 256 + c]
                                : p.curAtt[(size_t)(b0 + r) * 256 + (c - 256)];
            sAh[packA(r, c)] = (_Float16)v;
        }
        __syncthreads();
        gemm16(sAh, 512, p.projW, 256, p.projB, p.decin + (size_t)b0 * 256, 256, 0, wave, NW, lane);
        __syncthreads();

        // ---------------- decoder GRU 1 ----------------
        for (int idx = tid; idx < 16 * 256; idx += 1024)
            sAh[packA(idx >> 8, idx & 255)] = (_Float16)p.decin[(size_t)b0 * 256 + idx];
        __syncthreads();
        gemm16(sAh, 256, p.d1Wi, 768, p.d1Bi, p.gi + (size_t)b0 * 768, 768, 0, wave, NW, lane);
        __syncthreads();
        for (int idx = tid; idx < 16 * 256; idx += 1024)
            sAh[packA(idx >> 8, idx & 255)] = (_Float16)p.h1[(size_t)b0 * 256 + idx];
        __syncthreads();
        gemm16(sAh, 256, p.d1Wh, 768, p.d1Bh, p.gh + (size_t)b0 * 768, 768, 0, wave, NW, lane);
        __syncthreads();
        gru_combine(p.gi, p.gh, p.h1, b0, tid);
        __syncthreads();
        for (int idx = tid; idx < 16 * 256; idx += 1024) {
            size_t o = (size_t)b0 * 256 + idx;
            p.dcur[o] = p.h1[o] + p.decin[o];
        }
        __syncthreads();

        // ---------------- decoder GRU 2 ----------------
        for (int idx = tid; idx < 16 * 256; idx += 1024)
            sAh[packA(idx >> 8, idx & 255)] = (_Float16)p.dcur[(size_t)b0 * 256 + idx];
        __syncthreads();
        gemm16(sAh, 256, p.d2Wi, 768, p.d2Bi, p.gi + (size_t)b0 * 768, 768, 0, wave, NW, lane);
        __syncthreads();
        for (int idx = tid; idx < 16 * 256; idx += 1024)
            sAh[packA(idx >> 8, idx & 255)] = (_Float16)p.h2[(size_t)b0 * 256 + idx];
        __syncthreads();
        gemm16(sAh, 256, p.d2Wh, 768, p.d2Bh, p.gh + (size_t)b0 * 768, 768, 0, wave, NW, lane);
        __syncthreads();
        gru_combine(p.gi, p.gh, p.h2, b0, tid);
        __syncthreads();
        for (int idx = tid; idx < 16 * 256; idx += 1024) {
            size_t o = (size_t)b0 * 256 + idx;
            p.dcur[o] = p.h2[o] + p.dcur[o];
        }
        __syncthreads();

        // ---------------- outputs: mel (WMMA) + gate (VALU) ----------------
        for (int idx = tid; idx < 16 * 256; idx += 1024)
            sAh[packA(idx >> 8, idx & 255)] = (_Float16)p.dcur[(size_t)b0 * 256 + idx];
        __syncthreads();
        gemm16(sAh, 256, p.melW, DIN, p.melB,
               p.outMel + (size_t)b0 * TDEC * DIN + (size_t)t * DIN, (size_t)TDEC * DIN,
               0, wave, NW, lane);
        if (tid < 32) {
            int r = tid >> 1, g = tid & 1;
            int b = b0 + r;
            const float* drow = p.dcur + (size_t)b * 256;
            float acc = p.gateB[g];
            for (int j = 0; j < 256; ++j) acc += drow[j] * p.gateW[j * 2 + g];
            p.outGate[((size_t)b * TDEC + t) * 2 + g] = 1.f / (1.f + __expf(-acc));
        }
        __syncthreads();
    }
}

extern "C" void kernel_launch(void* const* d_in, const int* in_sizes, int n_in,
                              void* d_out, int out_size, void* d_ws, size_t ws_size,
                              hipStream_t stream)
{
    (void)in_sizes; (void)n_in; (void)out_size; (void)ws_size;
    const float* enc    = (const float*)d_in[0];
    const float* inp    = (const float*)d_in[1];
    const int*   memlen = (const int*)d_in[2];
    const float* preW1  = (const float*)d_in[3];
    const float* preB1  = (const float*)d_in[4];
    const float* preW2  = (const float*)d_in[5];
    const float* preB2  = (const float*)d_in[6];
    const float* memW   = (const float*)d_in[7];
    const float* qW     = (const float*)d_in[8];
    const float* vW     = (const float*)d_in[9];
    const float* attWi  = (const float*)d_in[10];
    const float* attWh  = (const float*)d_in[11];
    const float* attBi  = (const float*)d_in[12];
    const float* attBh  = (const float*)d_in[13];
    const float* projW  = (const float*)d_in[14];
    const float* projB  = (const float*)d_in[15];
    const float* d1Wi   = (const float*)d_in[16];
    const float* d1Wh   = (const float*)d_in[17];
    const float* d1Bi   = (const float*)d_in[18];
    const float* d1Bh   = (const float*)d_in[19];
    const float* d2Wi   = (const float*)d_in[20];
    const float* d2Wh   = (const float*)d_in[21];
    const float* d2Bi   = (const float*)d_in[22];
    const float* d2Bh   = (const float*)d_in[23];
    const float* melW   = (const float*)d_in[24];
    const float* melB   = (const float*)d_in[25];
    const float* gateW  = (const float*)d_in[26];
    const float* gateB  = (const float*)d_in[27];

    char* ws = (char*)d_ws;
    size_t off = 0;
    auto alloc = [&](size_t bytes) -> char* {
        off = (off + 255) & ~(size_t)255;
        char* r = ws + off;
        off += bytes;
        return r;
    };

    float*    pm      = (float*)   alloc((size_t)B_SZ * TENC * 256 * 4);
    _Float16* pMemW   = (_Float16*)alloc((size_t)256 * 256 * 2);
    _Float16* pPreW1  = (_Float16*)alloc((size_t)160 * 256 * 2);
    _Float16* pPreW2  = (_Float16*)alloc((size_t)256 * 128 * 2);
    _Float16* pQW     = (_Float16*)alloc((size_t)256 * 256 * 2);
    _Float16* pAttWi  = (_Float16*)alloc((size_t)384 * 768 * 2);
    _Float16* pAttWh  = (_Float16*)alloc((size_t)256 * 768 * 2);
    _Float16* pProjW  = (_Float16*)alloc((size_t)512 * 256 * 2);
    _Float16* pD1Wi   = (_Float16*)alloc((size_t)256 * 768 * 2);
    _Float16* pD1Wh   = (_Float16*)alloc((size_t)256 * 768 * 2);
    _Float16* pD2Wi   = (_Float16*)alloc((size_t)256 * 768 * 2);
    _Float16* pD2Wh   = (_Float16*)alloc((size_t)256 * 768 * 2);
    _Float16* pMelW   = (_Float16*)alloc((size_t)256 * 160 * 2);
    float* attH  = (float*)alloc((size_t)B_SZ * 256 * 4);
    float* h1    = (float*)alloc((size_t)B_SZ * 256 * 4);
    float* h2    = (float*)alloc((size_t)B_SZ * 256 * 4);
    float* curAtt= (float*)alloc((size_t)B_SZ * 256 * 4);
    float* gi    = (float*)alloc((size_t)B_SZ * 768 * 4);
    float* gh    = (float*)alloc((size_t)B_SZ * 768 * 4);
    float* q     = (float*)alloc((size_t)B_SZ * 256 * 4);
    float* t256  = (float*)alloc((size_t)B_SZ * 256 * 4);
    float* t128  = (float*)alloc((size_t)B_SZ * 128 * 4);
    float* decin = (float*)alloc((size_t)B_SZ * 256 * 4);
    float* dcur  = (float*)alloc((size_t)B_SZ * 256 * 4);

    auto pack = [&](const float* src, _Float16* dst, int K, int N) {
        int total = K * N;
        pack_kernel<<<(total + 255) / 256, 256, 0, stream>>>(src, dst, K, N);
    };
    pack(memW,  pMemW,  256, 256);
    pack(preW1, pPreW1, 160, 256);
    pack(preW2, pPreW2, 256, 128);
    pack(qW,    pQW,    256, 256);
    pack(attWi, pAttWi, 384, 768);
    pack(attWh, pAttWh, 256, 768);
    pack(projW, pProjW, 512, 256);
    pack(d1Wi,  pD1Wi,  256, 768);
    pack(d1Wh,  pD1Wh,  256, 768);
    pack(d2Wi,  pD2Wi,  256, 768);
    pack(d2Wh,  pD2Wh,  256, 768);
    pack(melW,  pMelW,  256, 160);

    pm_kernel<<<4096, 256, 0, stream>>>(enc, pMemW, pm);

    float* outMel   = (float*)d_out;
    float* outAlign = outMel + (size_t)B_SZ * TDEC * DIN;
    float* outGate  = outAlign + (size_t)B_SZ * TDEC * TENC;

    DecP p;
    p.enc = enc; p.pm = pm; p.inp = inp; p.memlen = memlen; p.vW = vW;
    p.preW1 = pPreW1; p.preW2 = pPreW2; p.qW = pQW; p.attWi = pAttWi; p.attWh = pAttWh;
    p.projW = pProjW; p.d1Wi = pD1Wi; p.d1Wh = pD1Wh; p.d2Wi = pD2Wi; p.d2Wh = pD2Wh;
    p.melW = pMelW;
    p.preB1 = preB1; p.preB2 = preB2; p.attBi = attBi; p.attBh = attBh; p.projB = projB;
    p.d1Bi = d1Bi; p.d1Bh = d1Bh; p.d2Bi = d2Bi; p.d2Bh = d2Bh; p.melB = melB;
    p.gateW = gateW; p.gateB = gateB;
    p.attH = attH; p.h1 = h1; p.h2 = h2; p.curAtt = curAtt; p.gi = gi; p.gh = gh;
    p.q = q; p.t256 = t256; p.t128 = t128; p.decin = decin; p.dcur = dcur;
    p.outMel = outMel; p.outAlign = outAlign; p.outGate = outGate;

    decoder_kernel<<<4, 1024, 0, stream>>>(p);
}
